// ConsistentSelfAttentionProcessor_70300024701598
// MI455X (gfx1250) — matrix operand order repaired and checked by reading, not compiled, not run on hardware
//
#include <hip/hip_runtime.h>

typedef __attribute__((ext_vector_type(16))) _Float16 v16h;
typedef __attribute__((ext_vector_type(8)))  _Float16 v8h;
typedef __attribute__((ext_vector_type(8)))  float    v8f;

union V16 { v16h v; v8h h[2]; };

__device__ __forceinline__ v8f v8f_zero() {
  v8f z;
#pragma unroll
  for (int i = 0; i < 8; ++i) z[i] = 0.0f;
  return z;
}

__device__ __forceinline__ v8f wmma16(V16 a, V16 b, v8f c) {
  // D = A(16x32 f16) x B(32x16 f16) + C(16x16 f32)
  return __builtin_amdgcn_wmma_f32_16x16x32_f16(false, a.v, false, b.v,
                                                (short)0, c, false, false);
}

// ---------------------------------------------------------------------------
// Conversions
// ---------------------------------------------------------------------------
__global__ void k_f32_to_f16(const float* __restrict__ x, _Float16* __restrict__ y, int n) {
  int i = blockIdx.x * blockDim.x + threadIdx.x;
  if (i < n) y[i] = (_Float16)x[i];
}

// wt[n*dim + k] = (f16) w[k*dim + n]   (coalesced writes)
__global__ void k_w_transpose_f16(const float* __restrict__ w, _Float16* __restrict__ wt, int dim) {
  int i = blockIdx.x * blockDim.x + threadIdx.x;
  if (i >= dim * dim) return;
  int n = i / dim;
  int k = i - n * dim;
  wt[i] = (_Float16)w[(size_t)k * dim + n];
}

// ---------------------------------------------------------------------------
// f16 GEMM:  out = (A[MxK] @ B[KxN] + bias) * outScale (+ residual)
// B supplied transposed: BT[NxK].  128x128 block tile, 8 waves of 32x64.
// Double-buffered LDS, K-step 64 (2 WMMA k-windows per step).
// ---------------------------------------------------------------------------
#define KT  64  // K-step per stage
#define LDT 72  // padded LDS stride (halfs) for 64-wide K tiles

__global__ __launch_bounds__(256, 2) void k_gemm_f16(
    const _Float16* __restrict__ A,
    const _Float16* __restrict__ BT,
    const float* __restrict__ bias,
    const float* __restrict__ residual,
    _Float16* __restrict__ outH,
    float* __restrict__ outF,
    int M, int N, int K, float outScale)
{
  __shared__ _Float16 ldsA[2][128 * LDT];
  __shared__ _Float16 ldsB[2][128 * LDT];

  const int tid  = threadIdx.x;
  const int lane = tid & 31;
  const int wid  = tid >> 5;
  const int wm   = wid >> 1;   // 0..3
  const int wn   = wid & 1;    // 0..1
  const int half = lane >> 4;
  const int l16  = lane & 15;
  const int mBase = blockIdx.y * 128;
  const int nBase = blockIdx.x * 128;

  v8f acc[2][4];
#pragma unroll
  for (int i = 0; i < 2; ++i)
#pragma unroll
    for (int j = 0; j < 4; ++j) acc[i][j] = v8f_zero();

  const int koffA = half * 8;   // A fragment: lanes16-31 start at K=8
  const int koffB = half * 8;   // B fragment: lanes16-31 start at K=8

  // Stage one 128x64 tile of A and of BT into LDS buffer `buf`.
  auto stage = [&](int kt, int buf) {
#pragma unroll
    for (int i = 0; i < 4; ++i) {
      int c   = tid + i * 256;   // 0..1023 chunks of 8 halfs
      int row = c >> 3;          // 128 rows, 8 chunks per row
      int off = (c & 7) * 8;
      const _Float16* ga = &A [(size_t)(mBase + row) * K + kt + off];
      const _Float16* gb = &BT[(size_t)(nBase + row) * K + kt + off];
      *(v8h*)&ldsA[buf][row * LDT + off] = *(const v8h*)ga;
      *(v8h*)&ldsB[buf][row * LDT + off] = *(const v8h*)gb;
      // speculative prefetch of the tile after next (safely dropped OOB)
      __builtin_prefetch(ga + KT, 0, 1);
      __builtin_prefetch(gb + KT, 0, 1);
    }
  };

  auto compute = [&](int buf) {
#pragma unroll
    for (int w = 0; w < 2; ++w) {
      V16 af[2], bfr[4];
#pragma unroll
      for (int ms = 0; ms < 2; ++ms) {
        int row = wm * 32 + ms * 16 + l16;
        const _Float16* p = &ldsA[buf][row * LDT + w * 32 + koffA];
        af[ms].h[0] = *(const v8h*)p;
        af[ms].h[1] = *(const v8h*)(p + 16);
      }
#pragma unroll
      for (int ns = 0; ns < 4; ++ns) {
        int col = wn * 64 + ns * 16 + l16;
        const _Float16* p = &ldsB[buf][col * LDT + w * 32 + koffB];
        bfr[ns].h[0] = *(const v8h*)p;
        bfr[ns].h[1] = *(const v8h*)(p + 16);
      }
#pragma unroll
      for (int ms = 0; ms < 2; ++ms)
#pragma unroll
        for (int ns = 0; ns < 4; ++ns)
          acc[ms][ns] = wmma16(af[ms], bfr[ns], acc[ms][ns]);
    }
  };

  stage(0, 0);
  int buf = 0;
  for (int kt = 0; kt + KT < K; kt += KT) {
    __syncthreads();
    stage(kt + KT, buf ^ 1);   // fill other buffer while computing this one
    compute(buf);
    buf ^= 1;
  }
  __syncthreads();
  compute(buf);

  // Epilogue: C layout lane/vgpr -> (m = r + 8*half, n = l16)
#pragma unroll
  for (int ms = 0; ms < 2; ++ms) {
#pragma unroll
    for (int ns = 0; ns < 4; ++ns) {
      int col  = nBase + wn * 64 + ns * 16 + l16;
      float bv = bias[col];
#pragma unroll
      for (int r = 0; r < 8; ++r) {
        int row   = mBase + wm * 32 + ms * 16 + half * 8 + r;
        float v   = (acc[ms][ns][r] + bv) * outScale;
        size_t ix = (size_t)row * N + col;
        if (residual) v += residual[ix];
        if (outH) outH[ix] = (_Float16)v;
        else      outF[ix] = v;
      }
    }
  }
}

// ---------------------------------------------------------------------------
// Flash attention, "first-frame" KV: keys 0..S-1 from frame b*4, S..2S-1 own.
// Block: 128 threads = 4 waves, each wave owns 16 query rows.
// Per stage: 64 keys, double-buffered (K row-major, V transposed in LDS).
// ---------------------------------------------------------------------------
#define SD 72  // padded LDS stride (halfs) for 64-wide tiles

__global__ __launch_bounds__(128, 2) void k_attn(
    const _Float16* __restrict__ Qh,   // pre-scaled by 1/sqrt(hd)
    const _Float16* __restrict__ Kh,
    const _Float16* __restrict__ Vh,
    _Float16* __restrict__ Oh)
{
  __shared__ _Float16 ldsK [2][64 * SD];    // [key][hd]
  __shared__ _Float16 ldsVT[2][64 * SD];    // [hd][key]
  __shared__ _Float16 ldsP [4 * 16 * SD];   // per-wave P tile [row][key]

  const int tid  = threadIdx.x;
  const int lane = tid & 31;
  const int wid  = tid >> 5;
  const int half = lane >> 4;
  const int l16  = lane & 15;
  const int h    = blockIdx.y;
  const int bf   = blockIdx.z;
  const int b0   = (bf >> 2) << 2;          // reference frame of this batch

  const size_t Dm    = 1280;
  const size_t rowQ0 = (size_t)bf * 1024 + (size_t)blockIdx.x * 64 + (size_t)wid * 16;
  const size_t colH  = (size_t)h * 64;

  // Q fragments for the wave's 16 rows, 2 K-windows over hd (A layout)
  V16 qf[2];
  {
    size_t base = (rowQ0 + l16) * Dm + colH;
    int ko = half * 8;
#pragma unroll
    for (int w = 0; w < 2; ++w) {
      qf[w].h[0] = *(const v8h*)&Qh[base + w * 32 + ko];
      qf[w].h[1] = *(const v8h*)&Qh[base + w * 32 + ko + 16];
    }
  }

  v8f o[4];
#pragma unroll
  for (int t = 0; t < 4; ++t) o[t] = v8f_zero();
  float mrun[8], lrun[8];
#pragma unroll
  for (int r = 0; r < 8; ++r) { mrun[r] = -3.0e38f; lrun[r] = 0.0f; }

  _Float16* myP = &ldsP[wid * 16 * SD];

  // Stage 64 keys (stage index s over the 2048 combined keys)
  auto stageKV = [&](int s, int buf) {
    int gk = s * 64;
    size_t frameRow = ((gk < 1024) ? (size_t)b0 : (size_t)bf) * 1024 + (size_t)(gk & 1023);
#pragma unroll
    for (int i = 0; i < 4; ++i) {
      int c   = tid + i * 128;   // 0..511 chunks of 8 halfs
      int key = c >> 3;
      int off = (c & 7) * 8;
      size_t gix = (frameRow + key) * Dm + colH + off;
      v8h kv = *(const v8h*)&Kh[gix];
      *(v8h*)&ldsK[buf][key * SD + off] = kv;
      v8h vv = *(const v8h*)&Vh[gix];
#pragma unroll
      for (int j = 0; j < 8; ++j) ldsVT[buf][(off + j) * SD + key] = vv[j];
    }
  };

  stageKV(0, 0);
  int buf = 0;
  for (int s = 0; s < 32; ++s) {
    __syncthreads();
    if (s + 1 < 32) stageKV(s + 1, buf ^ 1);

    // Scores S = Q @ K^T : 4 key-tiles of 16, hd reduced in 2 WMMA steps
    v8f sc[4];
#pragma unroll
    for (int t = 0; t < 4; ++t) {
      v8f c0 = v8f_zero();
#pragma unroll
      for (int w = 0; w < 2; ++w) {
        V16 kf;
        int key = t * 16 + l16;
        int kb  = w * 32 + half * 8;
        const _Float16* p = &ldsK[buf][key * SD + kb];
        kf.h[0] = *(const v8h*)p;
        kf.h[1] = *(const v8h*)(p + 16);
        c0 = wmma16(qf[w], kf, c0);
      }
      sc[t] = c0;
    }

    // Online softmax: row stats live in C layout (m = r + 8*half)
    float mnew[8], fs[8], psum[8];
#pragma unroll
    for (int r = 0; r < 8; ++r) {
      float pm = fmaxf(fmaxf(sc[0][r], sc[1][r]), fmaxf(sc[2][r], sc[3][r]));
#pragma unroll
      for (int msk = 1; msk < 16; msk <<= 1)
        pm = fmaxf(pm, __shfl_xor(pm, msk, 32));
      mnew[r] = fmaxf(mrun[r], pm);
      fs[r]   = __expf(mrun[r] - mnew[r]);
      psum[r] = 0.0f;
    }
#pragma unroll
    for (int t = 0; t < 4; ++t) {
#pragma unroll
      for (int r = 0; r < 8; ++r) {
        float p = __expf(sc[t][r] - mnew[r]);
        psum[r] += p;
        myP[(half * 8 + r) * SD + t * 16 + l16] = (_Float16)p;
      }
    }
#pragma unroll
    for (int r = 0; r < 8; ++r) {
#pragma unroll
      for (int msk = 1; msk < 16; msk <<= 1)
        psum[r] += __shfl_xor(psum[r], msk, 32);
      lrun[r] = lrun[r] * fs[r] + psum[r];
      mrun[r] = mnew[r];
    }
#pragma unroll
    for (int t = 0; t < 4; ++t)
#pragma unroll
      for (int r = 0; r < 8; ++r) o[t][r] *= fs[r];

    // O += P @ V : P as A fragments (key = K-dim), V^T as B fragments
    V16 pf[2];
    {
      int ko = half * 8;
      size_t pb = (size_t)l16 * SD;
#pragma unroll
      for (int w = 0; w < 2; ++w) {
        pf[w].h[0] = *(const v8h*)&myP[pb + w * 32 + ko];
        pf[w].h[1] = *(const v8h*)&myP[pb + w * 32 + ko + 16];
      }
    }
#pragma unroll
    for (int t = 0; t < 4; ++t) {
#pragma unroll
      for (int w = 0; w < 2; ++w) {
        V16 vf;
        int col = t * 16 + l16;
        int kb  = w * 32 + half * 8;
        const _Float16* p = &ldsVT[buf][col * SD + kb];
        vf.h[0] = *(const v8h*)p;
        vf.h[1] = *(const v8h*)(p + 16);
        o[t] = wmma16(pf[w], vf, o[t]);
      }
    }
    buf ^= 1;
  }

  float inv[8];
#pragma unroll
  for (int r = 0; r < 8; ++r) inv[r] = 1.0f / lrun[r];
#pragma unroll
  for (int t = 0; t < 4; ++t)
#pragma unroll
    for (int r = 0; r < 8; ++r) {
      size_t row = rowQ0 + half * 8 + r;
      Oh[row * Dm + colH + t * 16 + l16] = (_Float16)(o[t][r] * inv[r]);
    }
}

// ---------------------------------------------------------------------------
// Launch
// ---------------------------------------------------------------------------
extern "C" void kernel_launch(void* const* d_in, const int* in_sizes, int n_in,
                              void* d_out, int out_size, void* d_ws, size_t ws_size,
                              hipStream_t stream) {
  (void)in_sizes; (void)n_in; (void)out_size; (void)ws_size;

  const float* X  = (const float*)d_in[0];
  const float* Wq = (const float*)d_in[1];
  const float* Wk = (const float*)d_in[2];
  const float* Wv = (const float*)d_in[3];
  const float* Wo = (const float*)d_in[4];
  const float* bq = (const float*)d_in[5];
  const float* bk = (const float*)d_in[6];
  const float* bv = (const float*)d_in[7];
  const float* bo = (const float*)d_in[8];
  float* out = (float*)d_out;

  const size_t NX = 8192ull * 1280ull;   // activations
  const size_t NW = 1280ull * 1280ull;   // weight matrix
  _Float16* Xh  = (_Float16*)d_ws;
  _Float16* WqT = Xh + NX;
  _Float16* WkT = WqT + NW;
  _Float16* WvT = WkT + NW;
  _Float16* WoT = WvT + NW;
  _Float16* Qh  = WoT + NW;
  _Float16* Kh  = Qh + NX;
  _Float16* Vh  = Kh + NX;
  _Float16* Ah  = Vh + NX;
  // total workspace: ~113 MB of f16

  k_f32_to_f16<<<(int)((NX + 255) / 256), 256, 0, stream>>>(X, Xh, (int)NX);
  int wgrid = (int)((NW + 255) / 256);
  k_w_transpose_f16<<<wgrid, 256, 0, stream>>>(Wq, WqT, 1280);
  k_w_transpose_f16<<<wgrid, 256, 0, stream>>>(Wk, WkT, 1280);
  k_w_transpose_f16<<<wgrid, 256, 0, stream>>>(Wv, WvT, 1280);
  k_w_transpose_f16<<<wgrid, 256, 0, stream>>>(Wo, WoT, 1280);

  dim3 gg(10, 64);  // N/128 x M/128
  // Q gets 1/sqrt(64)=0.125 folded in (applies to bias too, matching reference)
  k_gemm_f16<<<gg, 256, 0, stream>>>(Xh, WqT, bq, nullptr, Qh, nullptr, 8192, 1280, 1280, 0.125f);
  k_gemm_f16<<<gg, 256, 0, stream>>>(Xh, WkT, bk, nullptr, Kh, nullptr, 8192, 1280, 1280, 1.0f);
  k_gemm_f16<<<gg, 256, 0, stream>>>(Xh, WvT, bv, nullptr, Vh, nullptr, 8192, 1280, 1280, 1.0f);

  dim3 ga(16, 20, 8);  // q-blocks x heads x frames
  k_attn<<<ga, 128, 0, stream>>>(Qh, Kh, Vh, Ah);

  // Output projection + bias + fp32 residual -> fp32 out
  k_gemm_f16<<<gg, 256, 0, stream>>>(Ah, WoT, bo, X, nullptr, out, 8192, 1280, 1280, 1.0f);
}
